// DMMRLoss_71562745086384
// MI455X (gfx1250) — matrix operand order repaired and checked by reading, compiled, not run on hardware
//
#include <hip/hip_runtime.h>
#include <hip/hip_bf16.h>
#include <math.h>

#define PATCH  17
#define P3     4913          // 17^3
#define NPP    15
#define NPATCH 3375          // 15^3
#define SDIM   255
#define S2     65025         // 255*255
#define NF     64
#define KCH    32
#define NK     154           // ceil(4913/32)
#define NTILE  211           // ceil(3375/16)
#define FRAGE  (NK * 4 * 32 * 16)   // halfs per packed-W buffer = 315392

typedef __attribute__((ext_vector_type(16))) __bf16 v16bf;
typedef __attribute__((ext_vector_type(8)))  float  v8f;

// ---------------------------------------------------------------------------
// Pre-pack Wf/Wm (f32 [4913,64]) into WMMA B fragments in this code's contiguous k order (bf16):
// frag[(kc*4+t)*32 + lane][j] with n = t*16 + (lane&15),
// k = kc*32 + (lane<16 ? j : 16+j); k >= 4913 zero-padded.
// Also zeroes the global accumulators.
// ---------------------------------------------------------------------------
__global__ __launch_bounds__(256) void prep_kernel(const float* __restrict__ Wf,
                                                   const float* __restrict__ Wm,
                                                   unsigned short* __restrict__ wff,
                                                   unsigned short* __restrict__ wmf,
                                                   float* __restrict__ acc) {
  int gid = blockIdx.x * 256 + threadIdx.x;
  if (gid < 2) acc[gid] = 0.0f;
  int total = 2 * FRAGE;
  if (gid >= total) return;
  int mtx  = gid / FRAGE;
  int idx  = gid - mtx * FRAGE;
  int j    = idx & 15;
  int lane = (idx >> 4) & 31;
  int ct   = idx >> 9;                 // kc*4 + t
  int t    = ct & 3;
  int kc   = ct >> 2;
  int n = t * 16 + (lane & 15);
  int k = kc * KCH + ((lane < 16) ? j : 16 + j);
  const float* W = mtx ? Wm : Wf;
  float v = (k < P3) ? W[k * NF + n] : 0.0f;
  __bf16 b = (__bf16)v;
  unsigned short* dst = mtx ? wmf : wff;
  dst[idx] = __builtin_bit_cast(unsigned short, b);
}

// ---------------------------------------------------------------------------
// Main kernel: 1 workgroup = 16 patches (M=16, N=64). 4 waves split K
// (chunks of 32, strided by 4). Volume data is staged into double-buffered
// LDS via GLOBAL_LOAD_ASYNC_TO_LDS_B32 (ASYNCcnt pipelining); A fragments
// are built from LDS with f32->bf16 conversion; 8 bf16 WMMAs per chunk.
// ---------------------------------------------------------------------------
__global__ __launch_bounds__(128) void dmmr_kernel(const float* __restrict__ fixedv,
                                                   const float* __restrict__ movingv,
                                                   const unsigned short* __restrict__ wffu,
                                                   const unsigned short* __restrict__ wmfu,
                                                   float* __restrict__ acc) {
  __shared__ float sF32[4][2][16][KCH];   // per-wave double-buffered fixed staging
  __shared__ float sM32[4][2][16][KCH];   // per-wave double-buffered moving staging
  __shared__ float ffs[16][NF];
  __shared__ float mfs[16][NF];
  __shared__ int   zc[16];

  const int tid  = threadIdx.x;
  const int wave = tid >> 5;
  const int lane = tid & 31;
  const int T    = blockIdx.x;
  const int M    = lane & 15;
  const bool pvM = (T * 16 + M) < NPATCH;

  for (int i = tid; i < 16 * NF; i += 128) {
    (&ffs[0][0])[i] = 0.0f;
    (&mfs[0][0])[i] = 0.0f;
  }
  if (tid < 16) zc[tid] = 0;
  __syncthreads();

  // per-patch base offsets into the 255^3 volume (unrolled -> stays in VGPRs)
  int base[16];
#pragma unroll
  for (int i = 0; i < 16; ++i) {
    int p  = T * 16 + i;
    int pc = (p < NPATCH) ? p : 0;
    int pz = pc / 225;
    int py = (pc / 15) % 15;
    int px = pc % 15;
    base[i] = pz * (PATCH * S2) + py * (PATCH * SDIM) + px * PATCH;
  }

  const v16bf* bfp = (const v16bf*)wffu;
  const v16bf* bmp = (const v16bf*)wmfu;

  v8f accF[4], accM[4];
#pragma unroll
  for (int t = 0; t < 4; ++t) { accF[t] = (v8f)0.0f; accM[t] = (v8f)0.0f; }

  // Issue one chunk's async global->LDS copies (32 instrs, 128B each).
  // k is clamped in-bounds; zero-padded W fragments annihilate clamped lanes.
  auto issue = [&](int kcI, int bufI) {
    int k   = kcI * KCH + lane;
    int kcl = (k < P3) ? k : (P3 - 1);
    int z = kcl / 289;
    int r = kcl - z * 289;
    int y = r / 17;
    int x = r - y * 17;
    int zoff = z * S2 + y * SDIM + x;
    unsigned ldsF = (unsigned)(size_t)&sF32[wave][bufI][0][lane];
    unsigned ldsM = (unsigned)(size_t)&sM32[wave][bufI][0][lane];
#pragma unroll
    for (int i = 0; i < 16; ++i) {
      unsigned long long gf = (unsigned long long)(size_t)(fixedv  + (base[i] + zoff));
      unsigned long long gm = (unsigned long long)(size_t)(movingv + (base[i] + zoff));
      asm volatile("global_load_async_to_lds_b32 %0, %1, off"
                   :: "v"(ldsF + (unsigned)(i * (KCH * 4))), "v"(gf) : "memory");
      asm volatile("global_load_async_to_lds_b32 %0, %1, off"
                   :: "v"(ldsM + (unsigned)(i * (KCH * 4))), "v"(gm) : "memory");
    }
  };

  int zcnt = 0;
  int b = 0;
  issue(wave, 0);

  for (int kc = wave; kc < NK; kc += 4, b ^= 1) {
    const bool nxt = (kc + 4) < NK;
    if (nxt) issue(kc + 4, b ^ 1);        // pipeline next chunk behind ASYNCcnt

    // L2-resident packed-W fragments: load before the async wait so they overlap
    v16bf bFv[4], bMv[4];
#pragma unroll
    for (int t = 0; t < 4; ++t) {
      bFv[t] = bfp[(kc * 4 + t) * 32 + lane];
      bMv[t] = bmp[(kc * 4 + t) * 32 + lane];
    }

    if (nxt) asm volatile("s_wait_asynccnt 32" ::: "memory");
    else     asm volatile("s_wait_asynccnt 0"  ::: "memory");

    // A fragment, contiguous K layout (matches packed B):
    // lane<16: elems 0..15 -> K 0..15
    // lane>=16: elems 0..15 -> K 16..31
    v16bf aF, aM;
#pragma unroll
    for (int j = 0; j < 16; ++j) {
      int   kk = j + 16 * (lane >> 4);
      float f  = sF32[wave][b][M][kk];
      float m  = sM32[wave][b][M][kk];
      aF[j] = (__bf16)f;
      aM[j] = (__bf16)m;
      int kelem = kc * KCH + kk;
      if (pvM && (kelem < P3) && (f == 0.0f)) zcnt++;
    }
    __builtin_amdgcn_wave_barrier();      // reads done before next chunk's async writes land

#pragma unroll
    for (int t = 0; t < 4; ++t) {
      accF[t] = __builtin_amdgcn_wmma_f32_16x16x32_bf16(false, aF, false, bFv[t],
                                                        (short)0, accF[t], false, false);
      accM[t] = __builtin_amdgcn_wmma_f32_16x16x32_bf16(false, aM, false, bMv[t],
                                                        (short)0, accM[t], false, false);
    }
  }

  // cross-wave reduction of partial features (C/D layout: row = r + 8*(lane>=16))
  const int mrow = 8 * (lane >> 4);
#pragma unroll
  for (int t = 0; t < 4; ++t) {
#pragma unroll
    for (int r = 0; r < 8; ++r) {
      atomicAdd(&ffs[mrow + r][t * 16 + M], accF[t][r]);
      atomicAdd(&mfs[mrow + r][t * 16 + M], accM[t][r]);
    }
  }
  if (zcnt) atomicAdd(&zc[M], zcnt);
  __syncthreads();

  if (tid < 16) {
    int p = T * 16 + tid;
    if (p < NPATCH) {
      float s = 0.0f;
      for (int n = 0; n < NF; ++n) s += ffs[tid][n] * mfs[tid][n];
      float keep = ((float)zc[tid] <= 0.15f * (float)P3) ? 1.0f : 0.0f;
      if (keep > 0.0f) atomicAdd(&acc[0], tanhf(s));
      atomicAdd(&acc[1], keep);
    }
  }
}

__global__ void finish_kernel(const float* __restrict__ acc, float* __restrict__ out) {
  out[0] = acc[0] / fmaxf(acc[1], 1.0f);
}

extern "C" void kernel_launch(void* const* d_in, const int* in_sizes, int n_in,
                              void* d_out, int out_size, void* d_ws, size_t ws_size,
                              hipStream_t stream) {
  const float* fixedv  = (const float*)d_in[0];
  const float* movingv = (const float*)d_in[1];
  const float* Wf      = (const float*)d_in[2];
  const float* Wm      = (const float*)d_in[3];

  unsigned short* wff = (unsigned short*)d_ws;
  unsigned short* wmf = wff + FRAGE;
  float* acc = (float*)((char*)d_ws + (size_t)2 * FRAGE * sizeof(unsigned short));

  int prep_total = 2 * FRAGE;
  prep_kernel<<<(prep_total + 255) / 256, 256, 0, stream>>>(Wf, Wm, wff, wmf, acc);
  dmmr_kernel<<<NTILE, 128, 0, stream>>>(fixedv, movingv, wff, wmf, acc);
  finish_kernel<<<1, 1, 0, stream>>>(acc, (float*)d_out);
}
